// VGAE_3100966387958
// MI455X (gfx1250) — matrix operand order repaired and checked by reading, not compiled, not run on hardware
//
#include <hip/hip_runtime.h>
#include <hip/hip_bf16.h>

typedef __bf16 bf16;
typedef __attribute__((ext_vector_type(16))) __bf16 bf16x16;
typedef __attribute__((ext_vector_type(8)))  __bf16 bf16x8;
typedef __attribute__((ext_vector_type(8)))  float  f32x8;

#define NN 16384
#define EE 524288
#define FF 512
#define H1_ 256
#define H2_ 128

union V16u { bf16x16 v; bf16x8 h[2]; };

// A fragment: 16x32 bf16, row-major A[M][K]. Lane covers row (lane&15);
// half-wave selects K sub-blocks: VGPR0-3 -> K = half*8 + [0..7],
// VGPR4-7 -> K = 16 + half*8 + [0..7]. Two contiguous 16B loads.
__device__ inline bf16x16 load_a_frag(const bf16* __restrict__ A, int row, int K, int k0, int half) {
  const bf16x8* lo = (const bf16x8*)(A + (size_t)row * K + k0 + half * 8);
  const bf16x8* hi = (const bf16x8*)(A + (size_t)row * K + k0 + 16 + half * 8);
  V16u u; u.h[0] = *lo; u.h[1] = *hi; return u.v;
}

// B fragment: 32x16 bf16 of B = Bt^T, Bt row-major [Ncols][K]. Lane covers
// column (lane&15); K = half*8 + [0..7], then 16 + half*8 + [0..7].
__device__ inline bf16x16 load_b_frag(const bf16* __restrict__ Bt, int col, int K, int k0, int half) {
  const bf16x8* lo = (const bf16x8*)(Bt + (size_t)col * K + k0 + half * 8);
  const bf16x8* hi = (const bf16x8*)(Bt + (size_t)col * K + k0 + 16 + half * 8);
  V16u u; u.h[0] = *lo; u.h[1] = *hi; return u.v;
}

// C = A[M][K] @ Bt[Ncols][K]^T, fp32 out. Block: 256 thr = 8 waves (2x4),
// wave tile 32x32 (2x2 WMMA bf16 16x16x32, fp32 acc). Optional sigmoid.
__global__ __launch_bounds__(256) void vgae_gemm_bf16_wmma(
    const bf16* __restrict__ A, const bf16* __restrict__ Bt,
    float* __restrict__ C, int M, int Ncols, int K, int sigmoid_ep)
{
  const int lane = threadIdx.x & 31;
  const int wave = threadIdx.x >> 5;
  const int wrow = wave >> 2, wcol = wave & 3;
  const int half = lane >> 4;
  const int l15  = lane & 15;
  const int row0 = blockIdx.y * 64  + wrow * 32;
  const int col0 = blockIdx.x * 128 + wcol * 32;

  f32x8 acc00 = {}, acc01 = {}, acc10 = {}, acc11 = {};
  for (int k0 = 0; k0 < K; k0 += 32) {
    bf16x16 a0 = load_a_frag(A, row0 + l15,      K, k0, half);
    bf16x16 a1 = load_a_frag(A, row0 + 16 + l15, K, k0, half);
    bf16x16 b0 = load_b_frag(Bt, col0 + l15,      K, k0, half);
    bf16x16 b1 = load_b_frag(Bt, col0 + 16 + l15, K, k0, half);
    acc00 = __builtin_amdgcn_wmma_f32_16x16x32_bf16(false, a0, false, b0, (short)0, acc00, false, false);
    acc01 = __builtin_amdgcn_wmma_f32_16x16x32_bf16(false, a0, false, b1, (short)0, acc01, false, false);
    acc10 = __builtin_amdgcn_wmma_f32_16x16x32_bf16(false, a1, false, b0, (short)0, acc10, false, false);
    acc11 = __builtin_amdgcn_wmma_f32_16x16x32_bf16(false, a1, false, b1, (short)0, acc11, false, false);
  }

  const size_t ldc = (size_t)Ncols;
  for (int r = 0; r < 8; ++r) {
    const size_t r0 = (size_t)(row0 + half * 8 + r);
    const size_t r1 = r0 + 16;
    const size_t c0 = (size_t)(col0 + l15);
    const size_t c1 = c0 + 16;
    float v00 = acc00[r], v01 = acc01[r], v10 = acc10[r], v11 = acc11[r];
    if (sigmoid_ep) {
      v00 = 1.0f / (1.0f + __expf(-v00));
      v01 = 1.0f / (1.0f + __expf(-v01));
      v10 = 1.0f / (1.0f + __expf(-v10));
      v11 = 1.0f / (1.0f + __expf(-v11));
    }
    C[r0 * ldc + c0] = v00; C[r0 * ldc + c1] = v01;
    C[r1 * ldc + c0] = v10; C[r1 * ldc + c1] = v11;
  }
}

// ---- graph / elementwise kernels ----

__global__ void vgae_init_deg(float* deg) {
  int i = blockIdx.x * blockDim.x + threadIdx.x;
  if (i < NN) deg[i] = 1.0f;               // self-loop
}

__global__ void vgae_deg_scatter(const long long* __restrict__ ei, float* deg) {
  int e = blockIdx.x * blockDim.x + threadIdx.x;
  if (e < EE) atomicAdd(&deg[(int)ei[EE + e]], 1.0f);
}

__global__ void vgae_dinv(const float* __restrict__ deg, float* dinv) {
  int i = blockIdx.x * blockDim.x + threadIdx.x;
  if (i < NN) dinv[i] = rsqrtf(deg[i]);
}

__global__ void vgae_cvt_x(const float* __restrict__ x, bf16* __restrict__ xb) {
  size_t t = (size_t)blockIdx.x * blockDim.x + threadIdx.x;
  if (t < (size_t)NN * FF) xb[t] = (bf16)x[t];
}

__global__ void vgae_build_wgcT(const float* __restrict__ W, bf16* __restrict__ Wt) {
  int t = blockIdx.x * blockDim.x + threadIdx.x;    // 256*512
  if (t < H1_ * FF) {
    int n = t >> 9, k = t & 511;
    Wt[t] = (bf16)W[k * H1_ + n];
  }
}

__global__ void vgae_build_wcatT(const float* __restrict__ Wr_mu, const float* __restrict__ Wl_mu,
                                 const float* __restrict__ Wr_lv, const float* __restrict__ Wl_lv,
                                 bf16* __restrict__ Wt) {
  int t = blockIdx.x * blockDim.x + threadIdx.x;    // 256*512
  if (t < 256 * 512) {
    int n = t >> 9, k = t & 511;
    float v;
    if (n < H2_) v = (k < H1_) ? Wr_mu[k * H2_ + n] : Wl_mu[(k - H1_) * H2_ + n];
    else { int nn = n - H2_; v = (k < H1_) ? Wr_lv[k * H2_ + nn] : Wl_lv[(k - H1_) * H2_ + nn]; }
    Wt[t] = (bf16)v;
  }
}

__global__ void vgae_h1_init(const float* __restrict__ xw, const float* __restrict__ dinv,
                             const float* __restrict__ b_gc1,
                             float* __restrict__ h1, float* __restrict__ agg) {
  size_t t = (size_t)blockIdx.x * blockDim.x + threadIdx.x;   // N*256
  if (t < (size_t)NN * H1_) {
    int i = (int)(t >> 8), c = (int)(t & 255);
    float di = dinv[i];
    h1[t]  = xw[t] * di * di + b_gc1[c];
    agg[t] = 0.0f;
  }
}

__global__ void vgae_gcn_scatter(const long long* __restrict__ ei, const float* __restrict__ xw,
                                 const float* __restrict__ dinv, float* __restrict__ h1) {
  long long t = (long long)blockIdx.x * blockDim.x + threadIdx.x;  // E*64
  int e = (int)(t >> 6);
  if (e >= EE) return;
  int c = ((int)t & 63) << 2;
  int s = (int)ei[e], d = (int)ei[EE + e];
  float w = dinv[s] * dinv[d];
  float4 v = *(const float4*)(xw + (size_t)s * H1_ + c);
  float* out = h1 + (size_t)d * H1_ + c;
  atomicAdd(out + 0, v.x * w); atomicAdd(out + 1, v.y * w);
  atomicAdd(out + 2, v.z * w); atomicAdd(out + 3, v.w * w);
}

__global__ void vgae_agg_scatter(const long long* __restrict__ ei, const float* __restrict__ h1,
                                 float* __restrict__ agg) {
  long long t = (long long)blockIdx.x * blockDim.x + threadIdx.x;  // E*64
  int e = (int)(t >> 6);
  if (e >= EE) return;
  int c = ((int)t & 63) << 2;
  int s = (int)ei[e], d = (int)ei[EE + e];
  float4 v = *(const float4*)(h1 + (size_t)s * H1_ + c);
  float* out = agg + (size_t)d * H1_ + c;
  atomicAdd(out + 0, v.x); atomicAdd(out + 1, v.y);
  atomicAdd(out + 2, v.z); atomicAdd(out + 3, v.w);
}

__global__ void vgae_build_hcat(const float* __restrict__ agg, const float* __restrict__ h1,
                                bf16* __restrict__ hcat) {
  size_t t = (size_t)blockIdx.x * blockDim.x + threadIdx.x;   // N*512
  if (t < (size_t)NN * 512) {
    int i = (int)(t >> 9), k = (int)(t & 511);
    float v = (k < H1_) ? agg[(size_t)i * H1_ + k] : h1[(size_t)i * H1_ + (k - H1_)];
    hcat[t] = (bf16)v;
  }
}

__global__ void vgae_z_kernel(const float* __restrict__ mulv, const float* __restrict__ b_mu,
                              const float* __restrict__ b_lv, const float* __restrict__ eps,
                              float* __restrict__ out_z, float* __restrict__ out_mu,
                              float* __restrict__ out_lv, bf16* __restrict__ zb) {
  size_t t = (size_t)blockIdx.x * blockDim.x + threadIdx.x;   // N*128
  if (t < (size_t)NN * H2_) {
    int i = (int)(t >> 7), c = (int)(t & 127);
    float mu = mulv[(size_t)i * 256 + c]        + b_mu[c];
    float lv = mulv[(size_t)i * 256 + H2_ + c]  + b_lv[c];
    float z  = eps[t] * __expf(lv) + mu;
    out_z[t] = z; out_mu[t] = mu; out_lv[t] = lv;
    zb[t] = (bf16)z;
  }
}

extern "C" void kernel_launch(void* const* d_in, const int* in_sizes, int n_in,
                              void* d_out, int out_size, void* d_ws, size_t ws_size,
                              hipStream_t stream) {
  const float*     x     = (const float*)d_in[0];
  const long long* ei    = (const long long*)d_in[1];
  const float*     W_gc1 = (const float*)d_in[2];
  const float*     b_gc1 = (const float*)d_in[3];
  const float*     Wr_mu = (const float*)d_in[4];
  const float*     Wl_mu = (const float*)d_in[5];
  const float*     b_mu  = (const float*)d_in[6];
  const float*     Wr_lv = (const float*)d_in[7];
  const float*     Wl_lv = (const float*)d_in[8];
  const float*     b_lv  = (const float*)d_in[9];
  const float*     eps   = (const float*)d_in[10];

  char* ws = (char*)d_ws;
  // workspace layout (buffer reuse: xb->hcat, xw->mulv, agg->zb)
  float* deg   = (float*)(ws + 0);                      //  64 KB
  float* dinv  = (float*)(ws + 65536);                  //  64 KB
  bf16*  wgcT  = (bf16*) (ws + 131072);                 // 256 KB  [H1][F]
  bf16*  wcatT = (bf16*) (ws + 393216);                 // 256 KB  [256][512]
  bf16*  xb    = (bf16*) (ws + 655360);                 //  16 MB  [N][F] (reused as hcat)
  bf16*  hcat  = xb;                                    //         [N][512]
  float* xw    = (float*)(ws + 17432576);               //  16 MB  [N][H1] (reused as mulv)
  float* mulv  = xw;                                    //         [N][256] = [mu|lv]
  float* h1    = (float*)(ws + 34209792);               //  16 MB  [N][H1]
  float* agg   = (float*)(ws + 50987008);               //  16 MB  [N][H1]
  bf16*  zb    = (bf16*) (ws + 50987008);               //   4 MB  [N][H2] (reuses agg)

  float* o_adj = (float*)d_out;                         // [N][N]
  float* o_z   = o_adj + (size_t)NN * NN;               // [N][H2]
  float* o_mu  = o_z  + (size_t)NN * H2_;
  float* o_lv  = o_mu + (size_t)NN * H2_;

  // 1) degrees + symmetric norm
  vgae_init_deg<<<NN / 256, 256, 0, stream>>>(deg);
  vgae_deg_scatter<<<EE / 256, 256, 0, stream>>>(ei, deg);
  vgae_dinv<<<NN / 256, 256, 0, stream>>>(deg, dinv);

  // 2) xw = x @ W_gc1  (bf16 WMMA)
  vgae_cvt_x<<<(NN * FF) / 256, 256, 0, stream>>>(x, xb);
  vgae_build_wgcT<<<(H1_ * FF) / 256, 256, 0, stream>>>(W_gc1, wgcT);
  vgae_gemm_bf16_wmma<<<dim3(H1_ / 128, NN / 64), 256, 0, stream>>>(
      xb, wgcT, xw, NN, H1_, FF, 0);

  // 3) GCN aggregation: h1 = self + sum_e norm*xw[src] + bias ; agg = 0
  vgae_h1_init<<<(NN * H1_) / 256, 256, 0, stream>>>(xw, dinv, b_gc1, h1, agg);
  vgae_gcn_scatter<<<(EE * 64) / 256, 256, 0, stream>>>(ei, xw, dinv, h1);

  // 4) GraphConv aggregation: agg = segment_sum(h1[src] -> dst)
  vgae_agg_scatter<<<(EE * 64) / 256, 256, 0, stream>>>(ei, h1, agg);

  // 5) fused GraphConv GEMMs: [agg|h1] @ [[Wr];[Wl]] for mu and lv together
  vgae_build_hcat<<<(NN * 512) / 256, 256, 0, stream>>>(agg, h1, hcat);
  vgae_build_wcatT<<<(256 * 512) / 256, 256, 0, stream>>>(Wr_mu, Wl_mu, Wr_lv, Wl_lv, wcatT);
  vgae_gemm_bf16_wmma<<<dim3(256 / 128, NN / 64), 256, 0, stream>>>(
      hcat, wcatT, mulv, NN, 256, 512, 0);

  // 6) reparameterize + emit z/mu/logvar, bf16 copy of z for decoder
  vgae_z_kernel<<<(NN * H2_) / 256, 256, 0, stream>>>(
      mulv, b_mu, b_lv, eps, o_z, o_mu, o_lv, zb);

  // 7) adj = sigmoid(z @ z^T)  — HBM-store-bound, bf16 WMMA compute
  vgae_gemm_bf16_wmma<<<dim3(NN / 128, NN / 64), 256, 0, stream>>>(
      zb, zb, o_adj, NN, NN, H2_, 1);
}
